// EmbedRNNSequenceAvg_57947698757847
// MI455X (gfx1250) — compile-verified, matrix operand order checked by reading, not run on hardware
//
#include <hip/hip_runtime.h>

// Problem constants (match reference)
#define Hdim 512
#define H2   1024
#define H3   1536
#define Oo   10000
#define Ll   50
#define Tt   8
#define Bb   1024

typedef __attribute__((ext_vector_type(16))) __bf16 v16bf_t;
typedef __attribute__((ext_vector_type(8)))  float  v8f_t;

union FragAB { v16bf_t v; uint4 q[2]; };

__device__ __forceinline__ unsigned short f2bf(float f) {
    unsigned int u = __float_as_uint(f);
    u += 0x7fffu + ((u >> 16) & 1u);          // round-to-nearest-even
    return (unsigned short)(u >> 16);
}
__device__ __forceinline__ float bf2f(unsigned short h) {
    return __uint_as_float(((unsigned int)h) << 16);
}

// A (16x32, M=row) / B (32x16, N=weight-row for X@W^T) bf16 fragment loader.
// ISA 7.12.2: lanes 0-15 hold K={k0..k0+7, k0+16..k0+23}; lanes 16-31 hold
// K={k0+8..k0+15, k0+24..k0+31}; 2 bf16 per VGPR, low-to-high.
__device__ __forceinline__ v16bf_t load_frag(const unsigned short* rowbase, int k0, int lane) {
    const int half = (lane >> 4) & 1;
    FragAB f;
    f.q[0] = *(const uint4*)(rowbase + k0 + half * 8);
    f.q[1] = *(const uint4*)(rowbase + k0 + 16 + half * 8);
    return f.v;
}

__device__ __forceinline__ v8f_t wmma_bf16(v16bf_t a, v16bf_t b, v8f_t c) {
    return __builtin_amdgcn_wmma_f32_16x16x32_bf16(false, a, false, b, (short)0, c, false, false);
}

// ---------------------------------------------------------------------------
// f32 -> bf16 conversion
__global__ __launch_bounds__(256) void f32_to_bf16_kernel(const float* __restrict__ src,
                                                          unsigned short* __restrict__ dst, int n) {
    int i = blockIdx.x * 256 + threadIdx.x;
    if (i < n) dst[i] = f2bf(src[i]);
}

// Embedding gather + bf16 convert: one block per token
__global__ __launch_bounds__(256) void gather_embed_kernel(const int* __restrict__ toks,
                                                           const float* __restrict__ embed,
                                                           unsigned short* __restrict__ out) {
    int t = blockIdx.x;
    const float* src = embed + (long long)toks[t] * Hdim;
    unsigned short* dst = out + (long long)t * Hdim;
    for (int j = threadIdx.x; j < Hdim; j += 256) dst[j] = f2bf(src[j]);
}

// ---------------------------------------------------------------------------
// Fused GRU step: h' = (1-z)*n + z*h with gates from two bf16 WMMA GEMMs.
// Block = 64(batch) x 64(hidden) tile; 8 waves; wave w -> row-strip (w&3),
// col-strips {2*(w>>2), 2*(w>>2)+1}. r/z kept in registers across gate passes
// (identical D-layout lane mapping in every pass).
__global__ __launch_bounds__(256) void gru_step_kernel(
    const unsigned short* __restrict__ x_bf16,   // [B,H] input this step
    const unsigned short* __restrict__ h_bf16,   // [B,H]
    const float*          __restrict__ h_f32,    // [B,H]
    const unsigned short* __restrict__ Wih,      // [3H,H] bf16
    const unsigned short* __restrict__ Whh,      // [3H,H] bf16
    const float* __restrict__ b_ih,
    const float* __restrict__ b_hh,
    float*          __restrict__ hn_f32,         // [B,H]
    unsigned short* __restrict__ hn_bf16,        // [B,H]
    unsigned short* __restrict__ enc_out_t,      // [B,H] masked encoder out (or null)
    const int* __restrict__ seq_lens, int tstep,
    float* __restrict__ pooled, float pool_scale)
{
    const int lane = threadIdx.x & 31;
    const int wave = threadIdx.x >> 5;
    const int rs   = wave & 3;
    const int ch   = wave >> 2;
    const int r0   = blockIdx.x * 64;
    const int c0   = blockIdx.y * 64;

    const int arow = r0 + rs * 16 + (lane & 15);
    const unsigned short* xrow = x_bf16 + (size_t)arow * Hdim;
    const unsigned short* hrow = h_bf16 + (size_t)arow * Hdim;

    const int wc0 = c0 + (ch * 2 + 0) * 16 + (lane & 15);
    const int wc1 = c0 + (ch * 2 + 1) * 16 + (lane & 15);

    float rz[2][2][8];  // [gate r/z][tile][row-i]

    for (int g = 0; g < 2; ++g) {   // gates r (0) and z (1): need gi+gh sum only
        const unsigned short* wi0 = Wih + (size_t)(g * Hdim + wc0) * Hdim;
        const unsigned short* wi1 = Wih + (size_t)(g * Hdim + wc1) * Hdim;
        const unsigned short* wh0 = Whh + (size_t)(g * Hdim + wc0) * Hdim;
        const unsigned short* wh1 = Whh + (size_t)(g * Hdim + wc1) * Hdim;
        v8f_t a0 = {}, a1 = {};
        for (int k0 = 0; k0 < Hdim; k0 += 32) {
            __builtin_prefetch(wi0 + k0 + 64, 0, 1);
            v16bf_t ax = load_frag(xrow, k0, lane);
            v16bf_t ah = load_frag(hrow, k0, lane);
            a0 = wmma_bf16(ax, load_frag(wi0, k0, lane), a0);
            a0 = wmma_bf16(ah, load_frag(wh0, k0, lane), a0);
            a1 = wmma_bf16(ax, load_frag(wi1, k0, lane), a1);
            a1 = wmma_bf16(ah, load_frag(wh1, k0, lane), a1);
        }
        float bb0 = b_ih[g * Hdim + wc0] + b_hh[g * Hdim + wc0];
        float bb1 = b_ih[g * Hdim + wc1] + b_hh[g * Hdim + wc1];
        for (int i = 0; i < 8; ++i) {
            rz[g][0][i] = 1.f / (1.f + expf(-(a0[i] + bb0)));
            rz[g][1][i] = 1.f / (1.f + expf(-(a1[i] + bb1)));
        }
    }

    // gate n: keep gi and gh separate (n = tanh(in + r*hn))
    {
        const unsigned short* wi0 = Wih + (size_t)(2 * Hdim + wc0) * Hdim;
        const unsigned short* wi1 = Wih + (size_t)(2 * Hdim + wc1) * Hdim;
        const unsigned short* wh0 = Whh + (size_t)(2 * Hdim + wc0) * Hdim;
        const unsigned short* wh1 = Whh + (size_t)(2 * Hdim + wc1) * Hdim;
        v8f_t ai0 = {}, ai1 = {}, ah0 = {}, ah1 = {};
        for (int k0 = 0; k0 < Hdim; k0 += 32) {
            v16bf_t ax = load_frag(xrow, k0, lane);
            v16bf_t ah = load_frag(hrow, k0, lane);
            ai0 = wmma_bf16(ax, load_frag(wi0, k0, lane), ai0);
            ai1 = wmma_bf16(ax, load_frag(wi1, k0, lane), ai1);
            ah0 = wmma_bf16(ah, load_frag(wh0, k0, lane), ah0);
            ah1 = wmma_bf16(ah, load_frag(wh1, k0, lane), ah1);
        }
        float bi0 = b_ih[2 * Hdim + wc0], bh0 = b_hh[2 * Hdim + wc0];
        float bi1 = b_ih[2 * Hdim + wc1], bh1 = b_hh[2 * Hdim + wc1];
        const int rbase = r0 + rs * 16 + ((lane < 16) ? 0 : 8);
        for (int i = 0; i < 8; ++i) {
            int row = rbase + i;
            // tile 0
            {
                float nn = tanhf((ai0[i] + bi0) + rz[0][0][i] * (ah0[i] + bh0));
                float z  = rz[1][0][i];
                size_t idx = (size_t)row * Hdim + wc0;
                float hnew = (1.f - z) * nn + z * h_f32[idx];
                hn_f32[idx]  = hnew;
                hn_bf16[idx] = f2bf(hnew);
                if (enc_out_t) enc_out_t[idx] = (tstep < seq_lens[row]) ? f2bf(hnew) : (unsigned short)0;
                if (pooled)    pooled[idx] += hnew * pool_scale;
            }
            // tile 1
            {
                float nn = tanhf((ai1[i] + bi1) + rz[0][1][i] * (ah1[i] + bh1));
                float z  = rz[1][1][i];
                size_t idx = (size_t)row * Hdim + wc1;
                float hnew = (1.f - z) * nn + z * h_f32[idx];
                hn_f32[idx]  = hnew;
                hn_bf16[idx] = f2bf(hnew);
                if (enc_out_t) enc_out_t[idx] = (tstep < seq_lens[row]) ? f2bf(hnew) : (unsigned short)0;
                if (pooled)    pooled[idx] += hnew * pool_scale;
            }
        }
    }
}

// ---------------------------------------------------------------------------
// Attention: scores[b,l] = [x,h] . attn_W[l,:] + attn_b[l]; softmax over L=50.
// One block per batch row; tiny K=1024 dot products done on VALU.
__global__ __launch_bounds__(256) void attn_scores_softmax_kernel(
    const unsigned short* __restrict__ x_bf16,  // [B,H] at step t
    const float* __restrict__ h_f32,            // [B,H]
    const float* __restrict__ attn_W,           // [L,2H]
    const float* __restrict__ attn_b,           // [L]
    float* __restrict__ aw_out)                 // [B,L]
{
    __shared__ float sS[Ll];
    int b = blockIdx.x;
    int lane = threadIdx.x & 31, wave = threadIdx.x >> 5;
    const unsigned short* xb = x_bf16 + (size_t)b * Hdim;
    const float*          hb = h_f32  + (size_t)b * Hdim;
    for (int l = wave; l < Ll; l += 8) {
        const float* wl = attn_W + (size_t)l * H2;
        float s = 0.f;
        for (int k = lane; k < Hdim; k += 32)
            s += bf2f(xb[k]) * wl[k] + hb[k] * wl[Hdim + k];
        for (int off = 16; off > 0; off >>= 1) s += __shfl_down(s, off, 32);
        if (lane == 0) sS[l] = s + attn_b[l];
    }
    __syncthreads();
    float mx = -1e30f;
    for (int l = 0; l < Ll; ++l) mx = fmaxf(mx, sS[l]);
    float den = 0.f;
    for (int l = 0; l < Ll; ++l) den += expf(sS[l] - mx);
    float inv = 1.f / den;
    if (threadIdx.x < Ll)
        aw_out[(size_t)b * Ll + threadIdx.x] = expf(sS[threadIdx.x] - mx) * inv;
}

// ctx[b,j] = sum_l aw[b,l] * enc_out[l,b,j]
__global__ __launch_bounds__(256) void ctx_kernel(const float* __restrict__ aw,          // [B,L]
                                                  const unsigned short* __restrict__ enc, // [L,B,H]
                                                  unsigned short* __restrict__ ctx_bf16) {
    int idx = blockIdx.x * 256 + threadIdx.x;  // b*H + j
    int b = idx >> 9, j = idx & (Hdim - 1);
    float s = 0.f;
    for (int l = 0; l < Ll; ++l)
        s += aw[(size_t)b * Ll + l] * bf2f(enc[((size_t)l * Bb + b) * Hdim + j]);
    ctx_bf16[idx] = f2bf(s);
}

// o = relu(concat(x,ctx) @ comb_W^T + comb_b)  -- WMMA, two-segment A (K=1024)
__global__ __launch_bounds__(256) void comb_relu_kernel(
    const unsigned short* __restrict__ x_bf16,   // [B,H]
    const unsigned short* __restrict__ ctx_bf16, // [B,H]
    const unsigned short* __restrict__ Wc,       // [H,2H] bf16
    const float* __restrict__ bc,                // [H]
    unsigned short* __restrict__ o_bf16)         // [B,H]
{
    const int lane = threadIdx.x & 31;
    const int wave = threadIdx.x >> 5;
    const int rs = wave & 3, ch = wave >> 2;
    const int r0 = blockIdx.x * 64, c0 = blockIdx.y * 64;
    const int arow = r0 + rs * 16 + (lane & 15);
    const int wc0 = c0 + (ch * 2 + 0) * 16 + (lane & 15);
    const int wc1 = c0 + (ch * 2 + 1) * 16 + (lane & 15);
    const unsigned short* w0 = Wc + (size_t)wc0 * H2;
    const unsigned short* w1 = Wc + (size_t)wc1 * H2;
    v8f_t a0 = {}, a1 = {};
    for (int k0 = 0; k0 < H2; k0 += 32) {
        // segment select: k<512 -> x, else ctx (base shifted so base+k0 lands right)
        const unsigned short* abase = (k0 < Hdim)
            ? (x_bf16 + (size_t)arow * Hdim)
            : (ctx_bf16 + (size_t)arow * Hdim - Hdim);
        v16bf_t av = load_frag(abase, k0, lane);
        a0 = wmma_bf16(av, load_frag(w0, k0, lane), a0);
        a1 = wmma_bf16(av, load_frag(w1, k0, lane), a1);
    }
    float b0 = bc[wc0], b1 = bc[wc1];
    const int rbase = r0 + rs * 16 + ((lane < 16) ? 0 : 8);
    for (int i = 0; i < 8; ++i) {
        int row = rbase + i;
        o_bf16[(size_t)row * Hdim + wc0] = f2bf(fmaxf(a0[i] + b0, 0.f));
        o_bf16[(size_t)row * Hdim + wc1] = f2bf(fmaxf(a1[i] + b1, 0.f));
    }
}

// logits = pooled @ fc_W^T + fc_b   (WMMA, N=10000 with edge guard)
__global__ __launch_bounds__(256) void fc_kernel(
    const unsigned short* __restrict__ pooled_bf16, // [B,H]
    const unsigned short* __restrict__ fcW,         // [O,H] bf16
    const float* __restrict__ fcb,                  // [O]
    float* __restrict__ logits)                     // [B,O]
{
    const int lane = threadIdx.x & 31;
    const int wave = threadIdx.x >> 5;
    const int rs = wave & 3, ch = wave >> 2;
    const int r0 = blockIdx.x * 64, c0 = blockIdx.y * 64;
    const int arow = r0 + rs * 16 + (lane & 15);
    const unsigned short* prow = pooled_bf16 + (size_t)arow * Hdim;
    const int wc0 = c0 + (ch * 2 + 0) * 16 + (lane & 15);
    const int wc1 = c0 + (ch * 2 + 1) * 16 + (lane & 15);
    const int wc0r = (wc0 < Oo) ? wc0 : (Oo - 1);
    const int wc1r = (wc1 < Oo) ? wc1 : (Oo - 1);
    const unsigned short* w0 = fcW + (size_t)wc0r * Hdim;
    const unsigned short* w1 = fcW + (size_t)wc1r * Hdim;
    v8f_t a0 = {}, a1 = {};
    for (int k0 = 0; k0 < Hdim; k0 += 32) {
        __builtin_prefetch(w0 + k0 + 64, 0, 1);
        v16bf_t av = load_frag(prow, k0, lane);
        a0 = wmma_bf16(av, load_frag(w0, k0, lane), a0);
        a1 = wmma_bf16(av, load_frag(w1, k0, lane), a1);
    }
    const int rbase = r0 + rs * 16 + ((lane < 16) ? 0 : 8);
    for (int i = 0; i < 8; ++i) {
        int row = rbase + i;
        if (wc0 < Oo) logits[(size_t)row * Oo + wc0] = a0[i] + fcb[wc0];
        if (wc1 < Oo) logits[(size_t)row * Oo + wc1] = a1[i] + fcb[wc1];
    }
}

// log_softmax over O=10000 per row
__global__ __launch_bounds__(256) void logsoftmax_kernel(const float* __restrict__ logits,
                                                         float* __restrict__ prob) {
    __shared__ float red[256];
    int b = blockIdx.x;
    const float* row = logits + (size_t)b * Oo;
    float mx = -1e30f;
    for (int j = threadIdx.x; j < Oo; j += 256) mx = fmaxf(mx, row[j]);
    red[threadIdx.x] = mx; __syncthreads();
    for (int s = 128; s > 0; s >>= 1) {
        if (threadIdx.x < s) red[threadIdx.x] = fmaxf(red[threadIdx.x], red[threadIdx.x + s]);
        __syncthreads();
    }
    mx = red[0]; __syncthreads();
    float sum = 0.f;
    for (int j = threadIdx.x; j < Oo; j += 256) sum += expf(row[j] - mx);
    red[threadIdx.x] = sum; __syncthreads();
    for (int s = 128; s > 0; s >>= 1) {
        if (threadIdx.x < s) red[threadIdx.x] += red[threadIdx.x + s];
        __syncthreads();
    }
    float lse = mx + logf(red[0]);
    float* out = prob + (size_t)b * Oo;
    for (int j = threadIdx.x; j < Oo; j += 256) out[j] = row[j] - lse;
}

// ---------------------------------------------------------------------------
extern "C" void kernel_launch(void* const* d_in, const int* in_sizes, int n_in,
                              void* d_out, int out_size, void* d_ws, size_t ws_size,
                              hipStream_t stream) {
    (void)in_sizes; (void)n_in; (void)out_size; (void)ws_size;

    const int*   sfs     = (const int*)d_in[0];
    const int*   lfs     = (const int*)d_in[1];
    const int*   lf_lens = (const int*)d_in[3];          // sf_lens (d_in[2]) unused by reference
    const float* embed   = (const float*)d_in[4];
    const float* W_ih_lf = (const float*)d_in[5];
    const float* W_hh_lf = (const float*)d_in[6];
    const float* b_ih_lf = (const float*)d_in[7];
    const float* b_hh_lf = (const float*)d_in[8];
    const float* W_ih_sf = (const float*)d_in[9];
    const float* W_hh_sf = (const float*)d_in[10];
    const float* b_ih_sf = (const float*)d_in[11];
    const float* b_hh_sf = (const float*)d_in[12];
    const float* attn_W  = (const float*)d_in[13];
    const float* attn_b  = (const float*)d_in[14];
    const float* comb_W  = (const float*)d_in[15];
    const float* comb_b  = (const float*)d_in[16];
    const float* fc_W    = (const float*)d_in[17];
    const float* fc_b    = (const float*)d_in[18];

    float* prob_out = (float*)d_out;                       // [B,O]
    float* attn_out = prob_out + (size_t)Bb * Oo;          // [T,B,L]

    // Workspace layout
    char* ws = (char*)d_ws;
    size_t off = 0;
    auto alloc = [&](size_t bytes) -> void* {
        void* p = ws + off;
        off = (off + bytes + 255) & ~(size_t)255;
        return p;
    };
    unsigned short* lf_emb   = (unsigned short*)alloc((size_t)Ll * Bb * Hdim * 2);
    unsigned short* sf_emb   = (unsigned short*)alloc((size_t)Tt * Bb * Hdim * 2);
    unsigned short* Wih_lf_b = (unsigned short*)alloc((size_t)H3 * Hdim * 2);
    unsigned short* Whh_lf_b = (unsigned short*)alloc((size_t)H3 * Hdim * 2);
    unsigned short* Wih_sf_b = (unsigned short*)alloc((size_t)H3 * Hdim * 2);
    unsigned short* Whh_sf_b = (unsigned short*)alloc((size_t)H3 * Hdim * 2);
    unsigned short* combW_b  = (unsigned short*)alloc((size_t)Hdim * H2 * 2);
    unsigned short* fcW_b    = (unsigned short*)alloc((size_t)Oo * Hdim * 2);
    float*          hA       = (float*)alloc((size_t)Bb * Hdim * 4);
    float*          hB       = (float*)alloc((size_t)Bb * Hdim * 4);
    unsigned short* hAb      = (unsigned short*)alloc((size_t)Bb * Hdim * 2);
    unsigned short* hBb      = (unsigned short*)alloc((size_t)Bb * Hdim * 2);
    unsigned short* enc_out  = (unsigned short*)alloc((size_t)Ll * Bb * Hdim * 2);
    unsigned short* ctxb     = (unsigned short*)alloc((size_t)Bb * Hdim * 2);
    unsigned short* ob       = (unsigned short*)alloc((size_t)Bb * Hdim * 2);
    float*          pooled   = (float*)alloc((size_t)Bb * Hdim * 4);
    unsigned short* pooledb  = (unsigned short*)alloc((size_t)Bb * Hdim * 2);
    float*          logits   = (float*)alloc((size_t)Bb * Oo * 4);

    dim3 blk(256);
    auto cvt = [&](const float* s, unsigned short* d, int n) {
        f32_to_bf16_kernel<<<(n + 255) / 256, blk, 0, stream>>>(s, d, n);
    };
    cvt(W_ih_lf, Wih_lf_b, H3 * Hdim);
    cvt(W_hh_lf, Whh_lf_b, H3 * Hdim);
    cvt(W_ih_sf, Wih_sf_b, H3 * Hdim);
    cvt(W_hh_sf, Whh_sf_b, H3 * Hdim);
    cvt(comb_W,  combW_b,  Hdim * H2);
    cvt(fc_W,    fcW_b,    Oo * Hdim);

    gather_embed_kernel<<<Ll * Bb, blk, 0, stream>>>(lfs, embed, lf_emb);
    gather_embed_kernel<<<Tt * Bb, blk, 0, stream>>>(sfs, embed, sf_emb);

    hipMemsetAsync(hA,     0, (size_t)Bb * Hdim * 4, stream);
    hipMemsetAsync(hAb,    0, (size_t)Bb * Hdim * 2, stream);
    hipMemsetAsync(pooled, 0, (size_t)Bb * Hdim * 4, stream);

    dim3 ggrid(Bb / 64, Hdim / 64);
    float* hc = hA; float* hn = hB;
    unsigned short* hcb = hAb; unsigned short* hnb = hBb;

    // Encoder: 50 sequential fused GRU steps
    for (int t = 0; t < Ll; ++t) {
        gru_step_kernel<<<ggrid, blk, 0, stream>>>(
            lf_emb + (size_t)t * Bb * Hdim, hcb, hc,
            Wih_lf_b, Whh_lf_b, b_ih_lf, b_hh_lf,
            hn, hnb, enc_out + (size_t)t * Bb * Hdim, lf_lens, t,
            nullptr, 0.f);
        { float* tf = hc; hc = hn; hn = tf; }
        { unsigned short* tu = hcb; hcb = hnb; hnb = tu; }
    }

    // Decoder: reset hidden, 8 steps of attention + comb + GRU (+ pooled mean)
    hipMemsetAsync(hc,  0, (size_t)Bb * Hdim * 4, stream);
    hipMemsetAsync(hcb, 0, (size_t)Bb * Hdim * 2, stream);
    for (int t = 0; t < Tt; ++t) {
        float* aw = attn_out + (size_t)t * Bb * Ll;
        attn_scores_softmax_kernel<<<Bb, blk, 0, stream>>>(
            sf_emb + (size_t)t * Bb * Hdim, hc, attn_W, attn_b, aw);
        ctx_kernel<<<(Bb * Hdim) / 256, blk, 0, stream>>>(aw, enc_out, ctxb);
        comb_relu_kernel<<<ggrid, blk, 0, stream>>>(
            sf_emb + (size_t)t * Bb * Hdim, ctxb, combW_b, comb_b, ob);
        gru_step_kernel<<<ggrid, blk, 0, stream>>>(
            ob, hcb, hc, Wih_sf_b, Whh_sf_b, b_ih_sf, b_hh_sf,
            hn, hnb, nullptr, nullptr, 0,
            pooled, 1.0f / Tt);
        { float* tf = hc; hc = hn; hn = tf; }
        { unsigned short* tu = hcb; hcb = hnb; hnb = tu; }
    }

    // Final FC + log_softmax
    cvt(pooled, pooledb, Bb * Hdim);
    dim3 fgrid(Bb / 64, (Oo + 63) / 64);
    fc_kernel<<<fgrid, blk, 0, stream>>>(pooledb, fcW_b, fc_b, logits);
    logsoftmax_kernel<<<Bb, blk, 0, stream>>>(logits, prob_out);
}
